// EqualizedModConv2d_52046413693165
// MI455X (gfx1250) — matrix operand order repaired and checked by reading, not compiled, not run on hardware
//
#include <hip/hip_runtime.h>
#include <hip/hip_bf16.h>

typedef __bf16 bf16_t;
typedef __attribute__((ext_vector_type(16))) __bf16 v16bf;
typedef __attribute__((ext_vector_type(8)))  __bf16 v8bf;
typedef __attribute__((ext_vector_type(8)))  float  v8f;
typedef int int4v __attribute__((ext_vector_type(4)));
typedef int i4_vs __attribute__((__vector_size__(16)));   // matches builtin param

// Problem constants
static constexpr int LATENT = 512;
static constexpr int CIN    = 512;
static constexpr int COUT   = 512;
static constexpr int BATCH  = 16;
static constexpr int H      = 64;
static constexpr int W      = 64;
static constexpr int OH     = 62;
static constexpr int OW     = 62;
static constexpr int KTOT   = CIN * 9;          // 4608 (k = i*9 + kh*3 + kw)

static constexpr float W_MUL_DENSE = 0.04419417382415922f;   // 512^-0.5
static constexpr float W_MUL_CONV  = 0.014731391274719739f;  // 4608^-0.5
static constexpr float W_MUL_CONV2 = 1.0f / 4608.0f;

// Tiling
static constexpr int BM = 256;      // Cout rows per block (grid.x = 2)
static constexpr int BN = 64;       // output columns (62 valid, padded)
static constexpr int BK = 64;       // K staged per barrier (2 x 32-K compute halves)
static constexpr int NIT = KTOT / BK;   // 72
static constexpr int APITCH = 72;   // 144B row pitch -> 16B-aligned LDS b128 ops
static constexpr int BPITCH = 72;

// Async global->LDS (CDNA5): direct LDS write, tracked by ASYNCcnt.
#if defined(__has_builtin)
#if __has_builtin(__builtin_amdgcn_global_load_async_to_lds_b128)
#define HAVE_ASYNC_LDS 1
#endif
#endif

#ifdef HAVE_ASYNC_LDS
__device__ __forceinline__ void async_copy16(const bf16_t* g, bf16_t* l) {
    __builtin_amdgcn_global_load_async_to_lds_b128(
        (__attribute__((address_space(1))) i4_vs*)(void*)(bf16_t*)g,
        (__attribute__((address_space(3))) i4_vs*)(void*)l, 0, 0);
}
#endif

// ---------------------------------------------------------------------------
__global__ void style_kernel(const float* __restrict__ y,
                             const float* __restrict__ dw,
                             const float* __restrict__ db,
                             float* __restrict__ s) {
    int idx = blockIdx.x * blockDim.x + threadIdx.x;      // 0..8191
    int b = idx >> 9, c = idx & 511;
    const float* yr = y + b * LATENT;
    const float* wr = dw + c * LATENT;
    float acc = 0.0f;
    for (int l = 0; l < LATENT; ++l) acc += yr[l] * wr[l];
    s[idx] = acc * W_MUL_DENSE + db[c];
}

__global__ void wsq_kernel(const float* __restrict__ w, float* __restrict__ wsq) {
    int idx = blockIdx.x * blockDim.x + threadIdx.x;      // o*512+i
    const float* p = w + (size_t)idx * 9;
    float a = 0.0f;
#pragma unroll
    for (int j = 0; j < 9; ++j) a += p[j] * p[j];
    wsq[idx] = a;
}

__global__ void demod_kernel(const float* __restrict__ s,
                             const float* __restrict__ wsq,
                             float* __restrict__ scale) {
    int idx = blockIdx.x * blockDim.x + threadIdx.x;      // b*512+o
    int b = idx >> 9, o = idx & 511;
    const float* sr = s + b * CIN;
    const float* wr = wsq + (size_t)o * CIN;
    float acc = 0.0f;
    for (int i = 0; i < CIN; ++i) { float sv = sr[i]; acc += sv * sv * wr[i]; }
    scale[idx] = rsqrtf(acc * W_MUL_CONV2 + 1e-8f) * W_MUL_CONV;
}

__global__ void modx_kernel(const float* __restrict__ x,
                            const float* __restrict__ s,
                            bf16_t* __restrict__ xs) {
    int idx = blockIdx.x * blockDim.x + threadIdx.x;
    int bi = idx >> 12;                                   // H*W = 4096
    xs[idx] = (bf16_t)(x[idx] * s[bi]);
}

__global__ void wcvt_kernel(const float* __restrict__ w, bf16_t* __restrict__ wbf) {
    int idx = blockIdx.x * blockDim.x + threadIdx.x;
    wbf[idx] = (bf16_t)w[idx];
}

// koff[k] = ci*H*W + kh*W + kw  with k = ci*9 + kh*3 + kw
__global__ void koff_kernel(int* __restrict__ koff) {
    int k = blockIdx.x * blockDim.x + threadIdx.x;        // 0..4607
    int ci = k / 9;
    int r  = k - 9 * ci;
    int kh = r / 3;
    int kw = r - 3 * kh;
    koff[k] = ci * (H * W) + kh * W + kw;
}

// ---------------------------------------------------------------------------
// Implicit-GEMM conv. Double-buffered LDS, async global->LDS A staging,
// 16 WMMA per wave per barrier (2 halves x 8).
// grid = (COUT/BM=2, OH, BATCH); block = 256 (8 waves).
__global__ __launch_bounds__(256)
void conv_kernel(const bf16_t* __restrict__ xs,
                 const bf16_t* __restrict__ wbf,
                 const float* __restrict__ scale,
                 const int* __restrict__ koff,
                 float* __restrict__ out) {
    __shared__ __align__(16) bf16_t As[2][BM][APITCH];   // [buf][cout-row][k]
    __shared__ __align__(16) bf16_t Bs[2][BN][BPITCH];   // [buf][n][k]

    const int b     = blockIdx.z;
    const int oy    = blockIdx.y;
    const int mBase = blockIdx.x * BM;
    const int tid   = threadIdx.x;
    const int lane  = tid & 31;
    const int wave  = tid >> 5;

    // A staging: one full 64-k row per thread (8 x 16B)
    const int arow = tid;
    const bf16_t* aRowPtr = wbf + (size_t)(mBase + arow) * KTOT;

    // B staging: thread owns column n = tid&63; two 8-k slices (one per half)
    const int bn   = tid & 63;
    const int kset = (tid >> 6) * 8;              // wave-uniform
    const int nc   = (bn < OW) ? bn : (OW - 1);   // clamp; cols 62,63 never stored
    const bf16_t* xslab = xs + (size_t)b * CIN * H * W;   // uniform base
    const int  nOff = oy * W + nc;                        // per-lane 32-bit offset
    const int4v* koffp = (const int4v*)koff;

    v8f acc[8] = {};

#ifdef HAVE_ASYNC_LDS
#define STAGE_A_COPY(S, D) async_copy16((S), (D));
#else
#define STAGE_A_COPY(S, D) *(v8bf*)(D) = *(const v8bf*)(S);
#endif
#define STAGE_A(T, BUF)                                                        \
    {                                                                          \
        const bf16_t* src = aRowPtr + (T) * BK;                                \
        bf16_t* dst = &As[BUF][arow][0];                                       \
        _Pragma("unroll")                                                      \
        for (int q = 0; q < 8; ++q) {                                          \
            STAGE_A_COPY(src + q * 8, dst + q * 8)                             \
        }                                                                      \
    }

    // k->offset table reads are wave-uniform (SGPR index via readfirstlane);
    // gathers use uniform base + unsigned 32-bit element offsets (GVS mode).
#define STAGE_B(T, BUF)                                                        \
    {                                                                          \
        _Pragma("unroll")                                                      \
        for (int h2 = 0; h2 < 2; ++h2) {                                       \
            int kb = __builtin_amdgcn_readfirstlane((T) * BK + h2 * 32 + kset);\
            int4v k0 = koffp[(kb >> 2)];                                       \
            int4v k1 = koffp[(kb >> 2) + 1];                                   \
            union { v8bf v; bf16_t e[8]; } bv;                                 \
            _Pragma("unroll")                                                  \
            for (int kk = 0; kk < 4; ++kk)                                     \
                bv.e[kk] = xslab[(unsigned)(k0[kk] + nOff)];                   \
            _Pragma("unroll")                                                  \
            for (int kk = 0; kk < 4; ++kk)                                     \
                bv.e[4 + kk] = xslab[(unsigned)(k1[kk] + nOff)];               \
            *(v8bf*)&Bs[BUF][bn][h2 * 32 + kset] = bv.v;                       \
        }                                                                      \
    }

    // prologue: stage tile 0 into buffer 0
    STAGE_A(0, 0)
    STAGE_B(0, 0)

    const int hk  = (lane < 16) ? 0 : 8;     // half-K select per ISA layouts
    const int l15 = lane & 15;

    for (int it = 0; it < NIT; ++it) {
        const int cur = it & 1;
#ifdef HAVE_ASYNC_LDS
        asm volatile("s_wait_asynccnt 0x0" ::: "memory");  // A-tile LDS writes done
#endif
        __syncthreads();
        if (it + 1 < NIT) {
            STAGE_A(it + 1, cur ^ 1)
            STAGE_B(it + 1, cur ^ 1)
        }

#pragma unroll
        for (int h = 0; h < 2; ++h) {
            // load ALL fragments for this 32-K half, then 8 back-to-back WMMAs
            union { v16bf v; v8bf h2[2]; } a0, a1, bfr[4];
            {
                const bf16_t* ap0 = &As[cur][wave * 32 + l15][h * 32 + hk];
                const bf16_t* ap1 = &As[cur][wave * 32 + 16 + l15][h * 32 + hk];
                a0.h2[0] = *(const v8bf*)(ap0);
                a0.h2[1] = *(const v8bf*)(ap0 + 16);
                a1.h2[0] = *(const v8bf*)(ap1);
                a1.h2[1] = *(const v8bf*)(ap1 + 16);
            }
#pragma unroll
            for (int j = 0; j < 4; ++j) {
                const bf16_t* bp = &Bs[cur][j * 16 + l15][h * 32 + hk];
                bfr[j].h2[0] = *(const v8bf*)(bp);
                bfr[j].h2[1] = *(const v8bf*)(bp + 16);
            }
#pragma unroll
            for (int j = 0; j < 4; ++j) {
                acc[j]     = __builtin_amdgcn_wmma_f32_16x16x32_bf16(
                    false, a0.v, false, bfr[j].v, (short)0, acc[j], false, false);
                acc[4 + j] = __builtin_amdgcn_wmma_f32_16x16x32_bf16(
                    false, a1.v, false, bfr[j].v, (short)0, acc[4 + j], false, false);
            }
        }
    }
#undef STAGE_A
#undef STAGE_A_COPY
#undef STAGE_B

    // epilogue: C/D layout: VGPR q, lanes 0-15 -> M=q, N=lane; lanes 16-31 -> M=8+q
    const size_t outB = (size_t)b * COUT * OH * OW + (size_t)oy * OW;
#pragma unroll
    for (int m = 0; m < 2; ++m) {
        const int oBase = mBase + wave * 32 + m * 16 + ((lane < 16) ? 0 : 8);
        float sc[8];
#pragma unroll
        for (int q = 0; q < 8; ++q) sc[q] = scale[b * COUT + oBase + q];
#pragma unroll
        for (int j = 0; j < 4; ++j) {
            int n = j * 16 + l15;
            if (n < OW) {
#pragma unroll
                for (int q = 0; q < 8; ++q)
                    out[outB + (size_t)(oBase + q) * OH * OW + n] =
                        acc[m * 4 + j][q] * sc[q];
            }
        }
    }
}

// ---------------------------------------------------------------------------
extern "C" void kernel_launch(void* const* d_in, const int* in_sizes, int n_in,
                              void* d_out, int out_size, void* d_ws, size_t ws_size,
                              hipStream_t stream) {
    const float* x  = (const float*)d_in[0];   // [16,512,64,64]
    const float* y  = (const float*)d_in[1];   // [16,512]
    const float* dw = (const float*)d_in[2];   // [512,512]
    const float* db = (const float*)d_in[3];   // [512]
    const float* w  = (const float*)d_in[4];   // [512,512,3,3]
    float* out = (float*)d_out;                // [16,512,62,62]

    // workspace layout (~73.0 MB)
    char* ws = (char*)d_ws;
    float*  s     = (float*)ws;                       // 8192 f32
    float*  scale = s + BATCH * COUT;                 // 8192 f32
    float*  wsq   = scale + BATCH * COUT;             // 262144 f32
    int*    koff  = (int*)(wsq + COUT * CIN);         // 4608 i32
    bf16_t* wbf   = (bf16_t*)(koff + KTOT);           // 2359296 bf16
    bf16_t* xs    = wbf + (size_t)COUT * KTOT;        // 33554432 bf16

    style_kernel<<<BATCH * COUT / 256, 256, 0, stream>>>(y, dw, db, s);
    wsq_kernel  <<<COUT * CIN / 256, 256, 0, stream>>>(w, wsq);
    demod_kernel<<<BATCH * COUT / 256, 256, 0, stream>>>(s, wsq, scale);
    modx_kernel <<<BATCH * CIN * H * W / 256, 256, 0, stream>>>(x, s, xs);
    wcvt_kernel <<<COUT * KTOT / 256, 256, 0, stream>>>(w, wbf);
    koff_kernel <<<KTOT / 256, 256, 0, stream>>>(koff);

    dim3 grid(COUT / BM, OH, BATCH);   // (2, 62, 16)
    conv_kernel<<<grid, 256, 0, stream>>>(xs, wbf, scale, koff, out);
}
